// VoxelResBackBone8xVoxelNeXt_70480413327749
// MI455X (gfx1250) — matrix operand order repaired and checked by reading, not compiled, not run on hardware
//
#include <hip/hip_runtime.h>
#include <stddef.h>

// ---------------------------------------------------------------------------
// CDNA5 (gfx1250) sparse-conv backbone: gather-GEMM via v_wmma_f32_16x16x32_bf16
// 32x64 double-buffered LDS A tile; each wave: 2 M-tiles x 2 N-tiles
// -> 8 WMMA per barrier, A and B fragments each reused by 2 WMMAs.
// ---------------------------------------------------------------------------

typedef __attribute__((ext_vector_type(16))) __bf16 v16bf;
typedef __attribute__((ext_vector_type(8)))  float  v8f;

__device__ __forceinline__ unsigned short f2bf(float f) {
  unsigned int u = __float_as_uint(f);
  u += 0x7fffu + ((u >> 16) & 1u);
  return (unsigned short)(u >> 16);
}
__device__ __forceinline__ unsigned int pk2bf(float a, float b) {
  return (unsigned int)f2bf(a) | ((unsigned int)f2bf(b) << 16);
}

union ABFrag { v16bf v; uint4 q[2]; };

__global__ __launch_bounds__(128)
void spconv_wmma_kernel(const float* __restrict__ feats,
                        const int* __restrict__ nbr,
                        int M, int Ktaps,
                        const unsigned short* __restrict__ wbf,
                        int Ci, int Cip, int Co,
                        const float* __restrict__ scale,
                        const float* __restrict__ shift,
                        const float* __restrict__ resid,
                        int do_relu,
                        float* __restrict__ out)
{
  // double-buffered 32 rows x 64 K bf16 tile (2 x 4KB)
  __shared__ __align__(16) unsigned short sA[2][32 * 64];

  const int tid  = threadIdx.x;
  const int wave = tid >> 5;
  const int lane = tid & 31;
  const int mt   = blockIdx.x;                 // 32-row output tile
  const int ctiles = Co >> 4;
  const int ct0 = blockIdx.y * 8 + wave;       // N-tile 1 for this wave
  const int ct1 = ct0 + 4;                     // N-tile 2 for this wave

  // staging role: 8 threads/row-pair, 8 bf16 (one b128) per row each
  const int sr = tid >> 3;                     // 0..15 (handles rows sr, sr+16)
  const int c0 = (tid & 7) * 8;

  const int s64   = (Cip > 64) ? 1 : 0;        // 64-K steps per tap
  const int nIter = Ktaps << s64;

  v8f acc00 = {}, acc01 = {}, acc10 = {}, acc11 = {};
  const int m  = lane & 15;                    // A: M = lane & 15
  const int kh = (lane < 16) ? 0 : 8;          // A: K half per lane group

  // ---- stage gathered 32x64 A tile for iteration `it` into buffer `p` ----
  auto stage = [&](int it, int p) {
    const int tap = it >> s64;
    const int kcb = (it & s64) << 6;           // K base within the tap
    #pragma unroll
    for (int h = 0; h < 2; ++h) {
      const int r  = sr + h * 16;
      const int gr = mt * 32 + r;
      int idx = -1;
      if (gr < M) idx = nbr[(size_t)gr * Ktaps + tap];
      uint4 pk;
      if (idx >= 0 && (kcb + c0 + 8) <= Ci) {
        const float4* sp = (const float4*)(feats + (size_t)idx * Ci + kcb + c0);
        const float4 f0 = sp[0];
        const float4 f1 = sp[1];
        pk.x = pk2bf(f0.x, f0.y); pk.y = pk2bf(f0.z, f0.w);
        pk.z = pk2bf(f1.x, f1.y); pk.w = pk2bf(f1.z, f1.w);
      } else {
        float fv[8];
        #pragma unroll
        for (int j = 0; j < 8; ++j) {
          const int k = kcb + c0 + j;
          fv[j] = (idx >= 0 && k < Ci) ? feats[(size_t)idx * Ci + k] : 0.0f;
        }
        pk.x = pk2bf(fv[0], fv[1]); pk.y = pk2bf(fv[2], fv[3]);
        pk.z = pk2bf(fv[4], fv[5]); pk.w = pk2bf(fv[6], fv[7]);
      }
      *(uint4*)(&sA[p][r * 64 + c0]) = pk;
    }
  };

  // ---- consume tile `p`: 2 K-sub x (2 M-tiles x 2 N-tiles) = 8 WMMA ----
  auto compute = [&](int it, int p) {
    const int tap = it >> s64;
    const int kcb = (it & s64) << 6;
    const int nsub = ((Cip - kcb) >= 64) ? 2 : 1;
    #pragma unroll
    for (int sub = 0; sub < 2; ++sub) {
      if (sub < nsub) {
        ABFrag a0, a1;
        const unsigned short* ap0 = &sA[p][m * 64 + sub * 32 + kh];
        a0.q[0] = *(const uint4*)(ap0);
        a0.q[1] = *(const uint4*)(ap0 + 16);
        const unsigned short* ap1 = ap0 + 16 * 64;       // rows 16..31
        a1.q[0] = *(const uint4*)(ap1);
        a1.q[1] = *(const uint4*)(ap1 + 16);
        const unsigned short* wrow =
            wbf + ((size_t)tap * Cip + kcb + sub * 32 + kh) * Co + (lane & 15);
        if (ct0 < ctiles) {
          ABFrag b;
          const unsigned short* wp = wrow + ct0 * 16;
          unsigned short* bs = (unsigned short*)b.q;
          #pragma unroll
          for (int s = 0; s < 16; ++s) bs[s] = wp[(size_t)(s < 8 ? s : s + 8) * Co];
          if (tap + 1 < Ktaps)
            __builtin_prefetch(wp + (size_t)Cip * Co, 0, 0);
          acc00 = __builtin_amdgcn_wmma_f32_16x16x32_bf16(
                      false, a0.v, false, b.v, (short)0, acc00, false, false);
          acc10 = __builtin_amdgcn_wmma_f32_16x16x32_bf16(
                      false, a1.v, false, b.v, (short)0, acc10, false, false);
        }
        if (ct1 < ctiles) {
          ABFrag b;
          const unsigned short* wp = wrow + ct1 * 16;
          unsigned short* bs = (unsigned short*)b.q;
          #pragma unroll
          for (int s = 0; s < 16; ++s) bs[s] = wp[(size_t)(s < 8 ? s : s + 8) * Co];
          acc01 = __builtin_amdgcn_wmma_f32_16x16x32_bf16(
                      false, a0.v, false, b.v, (short)0, acc01, false, false);
          acc11 = __builtin_amdgcn_wmma_f32_16x16x32_bf16(
                      false, a1.v, false, b.v, (short)0, acc11, false, false);
        }
      }
    }
  };

  // ---- software pipeline: one barrier per 64-K step ----
  stage(0, 0);
  __syncthreads();
  for (int it = 0; it < nIter; ++it) {
    const int p = it & 1;
    if (it + 1 < nIter) stage(it + 1, p ^ 1);
    compute(it, p);
    __syncthreads();
  }

  // ---- epilogue: fused scale/shift (+bias+BN), residual, ReLU ----
  const int nb = lane & 15;
  auto epi = [&](const v8f& acc, int h, int ct) {
    if (ct >= ctiles) return;
    const int n = ct * 16 + nb;
    const float s = scale[n];
    const float t = shift[n];
    #pragma unroll
    for (int j = 0; j < 8; ++j) {
      const int row = mt * 32 + h * 16 + j + ((lane < 16) ? 0 : 8);
      if (row < M) {
        float v = acc[j] * s + t;
        if (resid) v += resid[(size_t)row * Co + n];
        if (do_relu) v = fmaxf(v, 0.0f);
        out[(size_t)row * Co + n] = v;
      }
    }
  };
  epi(acc00, 0, ct0);
  epi(acc10, 1, ct0);
  epi(acc01, 0, ct1);
  epi(acc11, 1, ct1);
}

// f32 [K,Ci,Co] -> bf16 [K,Cip,Co], zero-padded Ci->Cip
__global__ void cvt_w_kernel(const float* __restrict__ w,
                             int K, int Ci, int Cip, int Co,
                             unsigned short* __restrict__ outw)
{
  const int t   = blockIdx.x * blockDim.x + threadIdx.x;
  const int tot = K * Cip * Co;
  if (t >= tot) return;
  const int co = t % Co;
  const int r  = t / Co;
  const int ci = r % Cip;
  const int k  = r / Cip;
  const float v = (ci < Ci) ? w[((size_t)k * Ci + ci) * Co + co] : 0.0f;
  outw[t] = f2bf(v);
}

// Fold conv-bias + batchnorm into scale/shift: y = x*scale + shift
__global__ void bn_prep_kernel(const float* __restrict__ g,
                               const float* __restrict__ b,
                               const float* __restrict__ m,
                               const float* __restrict__ v,
                               const float* __restrict__ bias, int C,
                               float* __restrict__ scale,
                               float* __restrict__ shift)
{
  const int i = blockIdx.x * blockDim.x + threadIdx.x;
  if (i >= C) return;
  const float s  = g[i] * rsqrtf(v[i] + 1e-3f);   // EPS = 0.001
  const float bb = bias ? bias[i] : 0.0f;
  scale[i] = s;
  shift[i] = (bb - m[i]) * s + b[i];
}

__global__ void zero_kernel(float* __restrict__ p, long long n)
{
  const long long t = (long long)blockIdx.x * blockDim.x + threadIdx.x;
  if (t < n) p[t] = 0.0f;
}

// segment-sum: bev[inv[i], c] += cat(x4,x5,x6)[i, c]
__global__ void bev_scatter_kernel(const float* __restrict__ x4,
                                   const float* __restrict__ x5,
                                   const float* __restrict__ x6,
                                   int n4, int n5, int n6,
                                   const int* __restrict__ inv,
                                   float* __restrict__ bev, int C)
{
  const long long tot = (long long)(n4 + n5 + n6) * C;
  const long long t   = (long long)blockIdx.x * blockDim.x + threadIdx.x;
  if (t >= tot) return;
  const int i = (int)(t / C);
  const int c = (int)(t % C);
  float v;
  if (i < n4)            v = x4[(size_t)i * C + c];
  else if (i < n4 + n5)  v = x5[(size_t)(i - n4) * C + c];
  else                   v = x6[(size_t)(i - n4 - n5) * C + c];
  unsafeAtomicAdd(&bev[(size_t)inv[i] * C + c], v);   // global_atomic_add_f32
}

// ---------------------------------------------------------------------------
// Host orchestration
// ---------------------------------------------------------------------------

namespace {

struct Prep { unsigned short* wb; float* sc; float* sh; int Cip; };

inline void* ws_take(char* ws, size_t& cur, size_t bytes) {
  cur = (cur + 255) & ~(size_t)255;
  void* p = ws + cur;
  cur += bytes;
  return p;
}

Prep prep_conv(const float* w, int K, int Ci, int Co,
               const float* g, const float* b, const float* m, const float* v,
               const float* bias, char* ws, size_t& cur, hipStream_t st)
{
  Prep p;
  p.Cip = (Ci + 31) & ~31;
  p.wb  = (unsigned short*)ws_take(ws, cur, (size_t)K * p.Cip * Co * 2);
  p.sc  = (float*)ws_take(ws, cur, (size_t)Co * 4);
  p.sh  = (float*)ws_take(ws, cur, (size_t)Co * 4);
  const int tot = K * p.Cip * Co;
  cvt_w_kernel<<<(tot + 255) / 256, 256, 0, st>>>(w, K, Ci, p.Cip, Co, p.wb);
  bn_prep_kernel<<<(Co + 63) / 64, 64, 0, st>>>(g, b, m, v, bias, Co, p.sc, p.sh);
  return p;
}

void launch_spconv(const float* in, const int* nbr, int M, int Ktaps,
                   const Prep& p, int Ci, int Co,
                   const float* resid, int relu, float* out, hipStream_t st)
{
  const int ctiles = Co / 16;
  dim3 grid((unsigned)((M + 31) / 32), (unsigned)((ctiles + 7) / 8));
  spconv_wmma_kernel<<<grid, 128, 0, st>>>(in, nbr, M, Ktaps, p.wb, Ci, p.Cip,
                                           Co, p.sc, p.sh, resid, relu, out);
}

} // namespace

extern "C" void kernel_launch(void* const* d_in, const int* in_sizes, int n_in,
                              void* d_out, int out_size, void* d_ws, size_t ws_size,
                              hipStream_t stream)
{
  (void)n_in; (void)out_size; (void)ws_size;
  auto F = [&](int i) { return (const float*)d_in[i]; };
  auto I = [&](int i) { return (const int*)d_in[i]; };

  // --- input index map (insertion-order flattening of setup_inputs) ---
  // 0 voxel_features, 1 voxel_coords
  // params: conv_input{w=2, bn g,b,m,v=3..6}
  //   res1 blocks @7,@19 (each: w1,b1,bn1(g,b,m,v),w2,b2,bn2(g,b,m,v))
  //   down2 w=31 bn=32..35 ; res2 @36,@48
  //   down3 w=60 bn=61..64 ; res3 @65,@77
  //   down4 w=89 bn=90..93 ; res4 @94,@106
  //   down5 w=118 bn=119..122 ; res5 @123,@135
  //   down6 w=147 bn=148..151 ; res6 @152,@164
  //   conv_out w=176 bn=177..180 ; shared w=181 b=182 bn=183..186
  // rb: sub1=187, down2=188,sub2=189, down3=190,sub3=191, down4=192,sub4=193,
  //     down5=194,sub5=195, down6=196,sub6=197, bev_inv=198, bev_base=199,
  //     out2d=200, sub2d=201
  static const int Cs[7]         = {0, 16, 32, 64, 128, 128, 128};
  static const int resBase[7][2] = {{0,0},{7,19},{36,48},{65,77},{94,106},{123,135},{152,164}};
  static const int downW[7]      = {0, 0, 31, 60, 89, 118, 147};
  static const int subIdx[7]     = {0, 187, 189, 191, 193, 195, 197};
  static const int downNbr[7]    = {0, 0, 188, 190, 192, 194, 196};

  int Ns[7];
  for (int s = 1; s <= 6; ++s) Ns[s] = in_sizes[subIdx[s]] / 27;
  const int U   = in_sizes[199];
  const int M2d = in_sizes[200] / 9;

  char* ws = (char*)d_ws;
  size_t cur = 0;

  // --- weight / BN preparation (bf16 weights, fused scale+shift) ---
  Prep Pin = prep_conv(F(2), 27, 5, 16, F(3), F(4), F(5), F(6), nullptr, ws, cur, stream);
  Prep Pres[7][2][2];
  for (int s = 1; s <= 6; ++s) {
    const int C = Cs[s];
    for (int blk = 0; blk < 2; ++blk) {
      const int bse = resBase[s][blk];
      Pres[s][blk][0] = prep_conv(F(bse),   27, C, C, F(bse+2), F(bse+3), F(bse+4),  F(bse+5),  F(bse+1), ws, cur, stream);
      Pres[s][blk][1] = prep_conv(F(bse+6), 27, C, C, F(bse+8), F(bse+9), F(bse+10), F(bse+11), F(bse+7), ws, cur, stream);
    }
  }
  Prep Pdown[7];
  for (int s = 2; s <= 6; ++s) {
    const int wi = downW[s];
    Pdown[s] = prep_conv(F(wi), 27, Cs[s-1], Cs[s], F(wi+1), F(wi+2), F(wi+3), F(wi+4), nullptr, ws, cur, stream);
  }
  Prep Pout = prep_conv(F(176), 9, 128, 128, F(177), F(178), F(179), F(180), nullptr, ws, cur, stream);
  Prep Psh  = prep_conv(F(181), 9, 128, 128, F(183), F(184), F(185), F(186), F(182), ws, cur, stream);

  // --- activation buffers (f32; whole net fits in the 192MB L2) ---
  float *A[7], *B[7], *T[7];
  for (int s = 1; s <= 6; ++s) {
    const size_t bytes = (size_t)Ns[s] * Cs[s] * 4;
    A[s] = (float*)ws_take(ws, cur, bytes);
    B[s] = (float*)ws_take(ws, cur, bytes);
    T[s] = (float*)ws_take(ws, cur, bytes);
  }
  float* bev = (float*)ws_take(ws, cur, (size_t)U * 128 * 4);
  float* Y1  = (float*)ws_take(ws, cur, (size_t)M2d * 128 * 4);

  // --- forward pass ---
  launch_spconv(F(0), I(subIdx[1]), Ns[1], 27, Pin, 5, 16, nullptr, 1, A[1], stream);

  for (int s = 1; s <= 6; ++s) {
    const int C = Cs[s];
    if (s >= 2)
      launch_spconv(A[s-1], I(downNbr[s]), Ns[s], 27, Pdown[s], Cs[s-1], C,
                    nullptr, 1, A[s], stream);
    const int* nb = I(subIdx[s]);
    // block 0:  T = relu(bn(conv1(A)));  B = relu(bn(conv2(T)) + A)
    launch_spconv(A[s], nb, Ns[s], 27, Pres[s][0][0], C, C, nullptr, 1, T[s], stream);
    launch_spconv(T[s], nb, Ns[s], 27, Pres[s][0][1], C, C, A[s],   1, B[s], stream);
    // block 1:  T = relu(bn(conv1(B)));  A = relu(bn(conv2(T)) + B)
    launch_spconv(B[s], nb, Ns[s], 27, Pres[s][1][0], C, C, nullptr, 1, T[s], stream);
    launch_spconv(T[s], nb, Ns[s], 27, Pres[s][1][1], C, C, B[s],   1, A[s], stream);
  }

  // --- BEV segment-sum over concat(stage4, stage5, stage6) ---
  {
    const long long nbev = (long long)U * 128;
    zero_kernel<<<(unsigned)((nbev + 255) / 256), 256, 0, stream>>>(bev, nbev);
    const long long tot = (long long)(Ns[4] + Ns[5] + Ns[6]) * 128;
    bev_scatter_kernel<<<(unsigned)((tot + 255) / 256), 256, 0, stream>>>(
        A[4], A[5], A[6], Ns[4], Ns[5], Ns[6], I(198), bev, 128);
  }

  // conv_out (9-tap 2D) then shared conv -> d_out
  launch_spconv(bev, I(200), M2d, 9, Pout, 128, 128, nullptr, 1, Y1, stream);
  launch_spconv(Y1,  I(201), M2d, 9, Psh,  128, 128, nullptr, 1, (float*)d_out, stream);
}
